// GRUNet_22608707846243
// MI455X (gfx1250) — compile-verified, matrix operand order checked by reading, not run on hardware
//
#include <hip/hip_runtime.h>
#include <hip/hip_bf16.h>

typedef __attribute__((ext_vector_type(16))) __bf16 v16bf;
typedef __attribute__((ext_vector_type(2)))  __bf16 v2bf;
typedef __attribute__((ext_vector_type(8)))  float  v8f;

#define HID_  19
#define FEAT_ 19
#define G3    57      // 3*HID
#define FC1_  10
#define B_TOT 8192
#define T_STEPS 168
#define NT    6       // gate-major tiles: {0,1}=z, {2,3}=r, {4,5}=h (units padded to 32)
#define ROWS  16      // batch rows per wave (WMMA M)

__device__ __forceinline__ float sigmoid_fast(float v) {
    return __builtin_amdgcn_rcpf(1.f + __expf(-v));
}
__device__ __forceinline__ float tanh_fast(float v) {
#if __has_builtin(__builtin_amdgcn_tanhf)
    return __builtin_amdgcn_tanhf(v);
#elif __has_builtin(__builtin_amdgcn_tanh_f32)
    return __builtin_amdgcn_tanh_f32(v);
#else
    return 1.f - 2.f * __builtin_amdgcn_rcpf(1.f + __expf(2.f * v));
#endif
}
__device__ __forceinline__ v8f splat8(float s) {
    v8f v;
    #pragma unroll
    for (int i = 0; i < 8; ++i) v[i] = s;
    return v;
}

// Single-wave workgroups: barriers are hardware S_NOPs; no inter-wave coupling.
__global__ __launch_bounds__(32)
void gru_fused_wmma(const float* __restrict__ x,
                    const float* __restrict__ kern,
                    const float* __restrict__ rec_kern,
                    const float* __restrict__ bias,
                    const float* __restrict__ W1,
                    const float* __restrict__ b1,
                    const float* __restrict__ W2,
                    const float* __restrict__ b2,
                    float* __restrict__ out)
{
    // h copy for the A_h rebuild, bf16, K-slot layout with the interleaved
    // permutation slot(2i)=unit i, slot(2i+1)=unit i+16: lane's two updated
    // units are adjacent -> one cvt_pk + one b32 store per row.
    __shared__ __bf16 H16[ROWS][32];   // pad slots stay 0 (1 KB)
    __shared__ float  Hf [ROWS][HID_]; // f32 h for the MLP head (written once)

    const int lane = threadIdx.x & 31;
    const int m    = lane & 15;      // A-row / C-column within tile
    const int hw   = lane >> 4;      // half-wave: K-octet / C row-block select
    const int b0   = blockIdx.x * ROWS;

    // ---- Resident bf16 B-fragments; x-GEMM K = features (direct),
    //      h-GEMM K = interleaved unit permutation ----
    v16bf Bk[NT], Br[NT];
    for (int j = 0; j < NT; ++j) {
        const int u   = m + 16 * (j & 1);
        const int col = (j >> 1) * HID_ + u;
        const bool uv = (u < HID_);
        for (int v = 0; v < 8; ++v) {
            const int k0 = (v >> 2) * 16 + hw * 8 + (v & 3) * 2;  // even
            float f0 = 0.f, f1 = 0.f, g0 = 0.f, g1 = 0.f;
            if (uv) {
                if (k0     < FEAT_) f0 = kern[k0 * G3 + col];
                if (k0 + 1 < FEAT_) f1 = kern[(k0 + 1) * G3 + col];
                const int ku0 = (k0 >> 1);        // unit for even slot, always < 16
                const int ku1 = (k0 >> 1) + 16;   // unit for odd slot
                g0 = rec_kern[ku0 * G3 + col];
                if (ku1 < HID_) g1 = rec_kern[ku1 * G3 + col];
            }
            Bk[j][2*v] = (__bf16)f0;  Bk[j][2*v+1] = (__bf16)f1;
            Br[j][2*v] = (__bf16)g0;  Br[j][2*v+1] = (__bf16)g1;
        }
    }

    // ---- Per-lane bias scalars ----
    float bz[2], brg[2], bxh[2], bhh[2];
    for (int g = 0; g < 2; ++g) {
        const int u = m + 16 * g;
        const bool uv = (u < HID_);
        bz [g] = uv ? bias[u]        + bias[G3 + u]        : 0.f;
        brg[g] = uv ? bias[HID_ + u] + bias[G3 + HID_ + u] : 0.f;
        bxh[g] = uv ? bias[2 * HID_ + u]                   : 0.f;
        bhh[g] = uv ? bias[G3 + 2 * HID_ + u]              : 0.f;
    }

    // ---- Hidden state: registers (gate distribution) + bf16 LDS (slot-major) ----
    v8f hreg[2];
    hreg[0] = splat8(0.f); hreg[1] = splat8(0.f);
    for (int i = lane; i < ROWS * 32; i += 32) H16[i / 32][i % 32] = (__bf16)0.f;
    __syncthreads();

    const float* xrow = x + ((size_t)(b0 + m) * T_STEPS) * FEAT_;

    for (int t = 0; t < T_STEPS; ++t) {
        const float* xr = xrow + (size_t)t * FEAT_;
        {   // stream prefetch: x advances only 76 B/step per row
            const int tp = (t + 8 < T_STEPS) ? t + 8 : T_STEPS - 1;
            __builtin_prefetch(xrow + (size_t)tp * FEAT_, 0, 0);
        }

        // ---- A fragments: x_t (global f32 -> bf16) and h (bf16 LDS, direct) ----
        v16bf Ax, Ah;
        #pragma unroll
        for (int v = 0; v < 8; ++v) {
            const int k0 = (v >> 2) * 16 + hw * 8 + (v & 3) * 2;  // k0+1 <= 31
            const float f0 = (k0     < FEAT_) ? __builtin_nontemporal_load(&xr[k0])     : 0.f;
            const float f1 = (k0 + 1 < FEAT_) ? __builtin_nontemporal_load(&xr[k0 + 1]) : 0.f;
            Ax[2*v] = (__bf16)f0;        Ax[2*v+1] = (__bf16)f1;
            Ah[2*v] = H16[m][k0];        Ah[2*v+1] = H16[m][k0 + 1];
        }

        // ---- 12x V_WMMA_F32_16X16X32_BF16 (C = inline 0) ----
        v8f accX[NT], accH[NT];
        const v8f cz = {};
        #pragma unroll
        for (int j = 0; j < NT; ++j) {
            accX[j] = __builtin_amdgcn_wmma_f32_16x16x32_bf16(
                false, Ax, false, Bk[j], (short)0, cz, false, false);
            accH[j] = __builtin_amdgcn_wmma_f32_16x16x32_bf16(
                false, Ah, false, Br[j], (short)0, cz, false, false);
        }

        // ---- Gate math: vector sums (pk_add on WMMA outputs), scalar trans ----
        v8f hn[2];
        #pragma unroll
        for (int g = 0; g < 2; ++g) {
            const v8f pz  = accX[g]     + accH[g]     + splat8(bz[g]);
            const v8f pr  = accX[2 + g] + accH[2 + g] + splat8(brg[g]);
            const v8f phx = accX[4 + g] + splat8(bxh[g]);
            const v8f phh = accH[4 + g] + splat8(bhh[g]);
            const v8f hv  = hreg[g];
            #pragma unroll
            for (int r = 0; r < 8; ++r) {
                const float z  = sigmoid_fast(pz[r]);
                const float rr = sigmoid_fast(pr[r]);
                const float hc = tanh_fast(phx[r] + rr * phh[r]);
                hn[g][r] = fmaf(z, hv[r] - hc, hc);
            }
            hreg[g] = hn[g];
        }
        // ---- Packed writeback: units (m, m+16) -> adjacent slots (2m, 2m+1) ----
        #pragma unroll
        for (int r = 0; r < 8; ++r) {
            v2bf p;
            p.x = (__bf16)hn[0][r];
            p.y = (__bf16)hn[1][r];
            *reinterpret_cast<v2bf*>(&H16[8 * hw + r][2 * m]) = p;
        }
        __syncthreads();   // compiler fence; hardware S_NOP (single-wave WG)
    }

    // ---- Stage final h (f32, valid units only) for the head ----
    #pragma unroll
    for (int g = 0; g < 2; ++g) {
        const int u = m + 16 * g;
        if (u < HID_)
            #pragma unroll
            for (int r = 0; r < 8; ++r)
                Hf[8 * hw + r][u] = hreg[g][r];
    }
    __syncthreads();

    // ---- MLP head: h -> relu(h@W1+b1) -> @W2+b2 ; one row per lane 0..15 ----
    if (hw == 0) {
        float hv[HID_];
        for (int k = 0; k < HID_; ++k) hv[k] = Hf[m][k];
        float o = b2[0];
        for (int j = 0; j < FC1_; ++j) {
            float a = b1[j];
            for (int k = 0; k < HID_; ++k) a += hv[k] * W1[k * FC1_ + j];
            o += (a > 0.f ? a : 0.f) * W2[j];
        }
        out[b0 + m] = o;
    }
}

extern "C" void kernel_launch(void* const* d_in, const int* in_sizes, int n_in,
                              void* d_out, int out_size, void* d_ws, size_t ws_size,
                              hipStream_t stream) {
    const float* x        = (const float*)d_in[0];
    const float* kern     = (const float*)d_in[1];
    const float* rec_kern = (const float*)d_in[2];
    const float* bias     = (const float*)d_in[3];
    const float* W1       = (const float*)d_in[4];
    const float* b1       = (const float*)d_in[5];
    const float* W2       = (const float*)d_in[6];
    const float* b2       = (const float*)d_in[7];
    float* out = (float*)d_out;

    const int blocks = B_TOT / ROWS;   // 512 single-wave workgroups
    gru_fused_wmma<<<blocks, 32, 0, stream>>>(
        x, kern, rec_kern, bias, W1, b1, W2, b2, out);
}
